// GPT_18296560681746
// MI455X (gfx1250) — compile-verified, matrix operand order checked by reading, not run on hardware
//
#include <hip/hip_runtime.h>

#define B_ 4
#define T_ 1024
#define E_ 768
#define H_ 12
#define L_ 6
#define V_ 50257
#define D_ 64
#define BT_ (B_ * T_)

#define BM 128
#define BN 128
#define BK 32

typedef _Float16 v16h __attribute__((ext_vector_type(16)));
typedef _Float16 v4h  __attribute__((ext_vector_type(4)));
typedef _Float16 h2   __attribute__((ext_vector_type(2)));
typedef float    v8f  __attribute__((ext_vector_type(8)));

// Generic pointer to LDS byte offset (low 32 bits of the shared aperture).
__device__ __forceinline__ unsigned lds_off(const void* p) {
  return (unsigned)(unsigned long long)p;
}

// Async 16B global->LDS copy (GVS mode: saddr base + 32-bit vgpr byte offset).
// Tracked by ASYNCcnt; completion enforced with s_wait_asynccnt.
__device__ __forceinline__ void cp_async16(unsigned lds, unsigned goff,
                                           unsigned long long base) {
  asm volatile("global_load_async_to_lds_b128 %0, %1, %2"
               :: "v"(lds), "v"(goff), "s"(base) : "memory");
}

// Build a WMMA 16-bit A/B fragment from an LDS row of 32 f16 values.
// Lanes 0-15 hold K={0..7,16..23}, lanes 16-31 hold K={8..15,24..31}.
__device__ __forceinline__ v16h ldsfrag(const _Float16* rowptr, int khalf) {
  v16h f;
#pragma unroll
  for (int v = 0; v < 8; ++v) {
    const int kb = 2 * v + ((v >= 4) ? 8 : 0) + 8 * khalf;
    h2 p = *(const h2*)(rowptr + kb);
    f[2 * v]     = p[0];
    f[2 * v + 1] = p[1];
  }
  return f;
}

__device__ __forceinline__ float redmax16(float v) {
#pragma unroll
  for (int m = 1; m < 16; m <<= 1) v = fmaxf(v, __shfl_xor(v, m, 16));
  return v;
}
__device__ __forceinline__ float redsum16(float v) {
#pragma unroll
  for (int m = 1; m < 16; m <<= 1) v += __shfl_xor(v, m, 16);
  return v;
}

// ---------------------------------------------------------------------------
// Double-buffered async WMMA GEMM: C[M,N] = A[M,K] @ W[N,K]^T
// 256 threads = 8 waves; tile 128x128, K-step 32; f16 operands streamed into
// LDS with global_load_async_to_lds_b128 (4 async instrs/wave/tile), pipelined
// with s_wait_asynccnt. Each wave owns a 16-row strip and 8 WMMA n-tiles.
// MODE: 0 = f32 out, 1 = f32 out + residual, 2 = f16 out (exact gelu),
//       3 = f16 out
// ---------------------------------------------------------------------------
template <int MODE>
__global__ __launch_bounds__(256) void gemm_wmma_async(
    const _Float16* __restrict__ A, const _Float16* __restrict__ W,
    const float* __restrict__ R, float* __restrict__ Cf,
    _Float16* __restrict__ Ch, int M, int N, int K) {
  __shared__ _Float16 As[2][BM][BK];
  __shared__ _Float16 Bs[2][BN][BK];
  const int tid   = threadIdx.x;
  const int lane  = tid & 31;
  const int wave  = tid >> 5;
  const int lrow  = lane & 15;
  const int khalf = lane >> 4;
  const int m0 = blockIdx.x * BM;
  const int n0 = blockIdx.y * BN;
  const int Nclamp = N - 1;

  const unsigned ldsA = lds_off(&As[0][0][0]);
  const unsigned ldsB = lds_off(&Bs[0][0][0]);
  const unsigned long long Abase = (unsigned long long)A;
  const unsigned long long Wbase = (unsigned long long)W;

  v8f zero = {};
  v8f acc[8] = {zero, zero, zero, zero, zero, zero, zero, zero};

  // Issue async copies for one 128x32 A tile + 128x32 B tile into buffer buf_.
  auto copy_tile = [&](int buf_, int k0_) {
    const unsigned boff = (unsigned)buf_ * (BM * BK * 2);
#pragma unroll
    for (int c = tid; c < (BM * BK) / 8; c += 256) {  // 512 chunks of 16B
      const int r  = c >> 2;
      const int cc = (c & 3) * 16;  // byte offset within the 64B row
      cp_async16(ldsA + boff + (unsigned)c * 16,
                 (unsigned)((m0 + r) * K) * 2 + (unsigned)k0_ * 2 + cc, Abase);
      int nr = n0 + r;
      if (nr > Nclamp) nr = Nclamp;  // ragged-N clamp (LM head)
      cp_async16(ldsB + boff + (unsigned)c * 16,
                 (unsigned)(nr * K) * 2 + (unsigned)k0_ * 2 + cc, Wbase);
    }
  };

  copy_tile(0, 0);
  int buf = 0;
  for (int k0 = 0; k0 < K; k0 += BK) {
    const bool more = (k0 + BK) < K;
    if (more) copy_tile(buf ^ 1, k0 + BK);
    // Wait for the tile we are about to consume (leave next tile in flight).
    if (more) asm volatile("s_wait_asynccnt 0x4" ::: "memory");
    else      asm volatile("s_wait_asynccnt 0x0" ::: "memory");
    __syncthreads();

    v16h af = ldsfrag(&As[buf][wave * 16 + lrow][0], khalf);
#pragma unroll
    for (int nt = 0; nt < 8; ++nt) {
      v16h bf = ldsfrag(&Bs[buf][nt * 16 + lrow][0], khalf);
      acc[nt] = __builtin_amdgcn_wmma_f32_16x16x32_f16(
          false, af, false, bf, (short)0, acc[nt], false, false);
    }
    __syncthreads();  // protect buf from being overwritten by next copies
    buf ^= 1;
  }

#pragma unroll
  for (int nt = 0; nt < 8; ++nt) {
    const int n = n0 + nt * 16 + lrow;
    if (n >= N) continue;
#pragma unroll
    for (int r = 0; r < 8; ++r) {
      const int m = m0 + wave * 16 + r + 8 * khalf;
      float v = acc[nt][r];
      if (MODE == 0) {
        Cf[(size_t)m * N + n] = v;
      } else if (MODE == 1) {
        Cf[(size_t)m * N + n] = v + R[(size_t)m * N + n];
      } else if (MODE == 2) {
        v = 0.5f * v * (1.0f + erff(v * 0.70710678118654752f));
        Ch[(size_t)m * N + n] = (_Float16)v;
      } else {
        Ch[(size_t)m * N + n] = (_Float16)v;
      }
    }
  }
}

// ---------------------------------------------------------------------------
// Flash attention over f16 qkv: block = (b, h, 64 query rows), 4 waves.
// S = QK^T via WMMA, online softmax (16-lane ds_bpermute reductions), P
// restaged through per-wave LDS into WMMA A-layout, O += P @ V via WMMA.
// ---------------------------------------------------------------------------
__global__ __launch_bounds__(128) void attn_flash(
    const _Float16* __restrict__ qkv,  // [B*T, 3E] f16
    _Float16* __restrict__ y) {        // [B*T, E]  f16
  __shared__ _Float16 Qs[64][64];
  __shared__ _Float16 Ks[32][64];     // [key][d]
  __shared__ _Float16 Vt[64][32];     // [d][key] (transposed at load)
  __shared__ _Float16 Ps[4][16][32];  // per-wave P staging

  const int qb = blockIdx.x;
  const int b  = blockIdx.y / H_;
  const int h  = blockIdx.y % H_;
  const int tid   = threadIdx.x;
  const int lane  = tid & 31;
  const int wave  = tid >> 5;
  const int lrow  = lane & 15;
  const int khalf = lane >> 4;
  const int q0 = qb * 64;
  const size_t rs = 3 * E_;
  const _Float16* qp = qkv + ((size_t)(b * T_ + q0)) * rs + h * D_;

  for (int i = tid; i < 64 * 16; i += 128) {  // 8B vector loads
    const int r = i >> 4, c = (i & 15) * 4;
    *(v4h*)&Qs[r][c] = *(const v4h*)&qp[(size_t)r * rs + c];
  }

  float mrow[8], lsum[8];
#pragma unroll
  for (int r = 0; r < 8; ++r) { mrow[r] = -__builtin_inff(); lsum[r] = 0.0f; }
  v8f zero = {};
  v8f o[4] = {zero, zero, zero, zero};

  const int nkb = 2 * qb + 2;
  for (int kb = 0; kb < nkb; ++kb) {
    __syncthreads();
    const _Float16* kp = qkv + ((size_t)(b * T_ + kb * 32)) * rs + E_ + h * D_;
    const _Float16* vp = kp + E_;
    for (int i = tid; i < 32 * 16; i += 128) {
      const int s = i >> 4, c = (i & 15) * 4;
      *(v4h*)&Ks[s][c] = *(const v4h*)&kp[(size_t)s * rs + c];
      v4h vv = *(const v4h*)&vp[(size_t)s * rs + c];
#pragma unroll
      for (int j = 0; j < 4; ++j) Vt[c + j][s] = vv[j];
    }
    __syncthreads();

    v8f sacc[2] = {zero, zero};
#pragma unroll
    for (int ks = 0; ks < 2; ++ks) {
      v16h af = ldsfrag(&Qs[wave * 16 + lrow][ks * 32], khalf);
#pragma unroll
      for (int nt = 0; nt < 2; ++nt) {
        v16h bf = ldsfrag(&Ks[nt * 16 + lrow][ks * 32], khalf);
        sacc[nt] = __builtin_amdgcn_wmma_f32_16x16x32_f16(
            false, af, false, bf, (short)0, sacc[nt], false, false);
      }
    }

#pragma unroll
    for (int r = 0; r < 8; ++r) {
      const int q  = q0 + wave * 16 + r + 8 * khalf;
      const int s0 = kb * 32 + lrow;
      const int s1 = s0 + 16;
      const float v0 = (s0 <= q) ? sacc[0][r] * 0.125f : -__builtin_inff();
      const float v1 = (s1 <= q) ? sacc[1][r] * 0.125f : -__builtin_inff();
      const float mnew  = fmaxf(mrow[r], redmax16(fmaxf(v0, v1)));
      const float p0    = expf(v0 - mnew);
      const float p1    = expf(v1 - mnew);
      const float alpha = expf(mrow[r] - mnew);
      lsum[r] = lsum[r] * alpha + redsum16(p0 + p1);
      mrow[r] = mnew;
#pragma unroll
      for (int nt = 0; nt < 4; ++nt) o[nt][r] *= alpha;
      Ps[wave][r + 8 * khalf][lrow]      = (_Float16)p0;
      Ps[wave][r + 8 * khalf][16 + lrow] = (_Float16)p1;
    }

    v16h pf = ldsfrag(&Ps[wave][lrow][0], khalf);
#pragma unroll
    for (int nt = 0; nt < 4; ++nt) {
      v16h vf = ldsfrag(&Vt[nt * 16 + lrow][0], khalf);
      o[nt] = __builtin_amdgcn_wmma_f32_16x16x32_f16(
          false, pf, false, vf, (short)0, o[nt], false, false);
    }
  }

  _Float16* yp = y + ((size_t)(b * T_ + q0)) * E_ + h * D_;
#pragma unroll
  for (int nt = 0; nt < 4; ++nt) {
    const int d = nt * 16 + lrow;
#pragma unroll
    for (int r = 0; r < 8; ++r) {
      const int qr = wave * 16 + r + 8 * khalf;
      yp[(size_t)qr * E_ + d] = (_Float16)(o[nt][r] / lsum[r]);
    }
  }
}

// ---------------------------------------------------------------------------
// Elementwise / reduction kernels
// ---------------------------------------------------------------------------
__global__ __launch_bounds__(256) void cvt_f32_to_f16(
    const float* __restrict__ s, _Float16* __restrict__ d, long n) {
  long i  = (long)blockIdx.x * 256 + threadIdx.x;
  long st = (long)gridDim.x * 256;
  for (; i < n; i += st) d[i] = (_Float16)s[i];
}

__global__ __launch_bounds__(256) void embed_kernel(
    const int* __restrict__ idx, const float* __restrict__ wte,
    const float* __restrict__ wpe, float* __restrict__ x) {
  const int row = blockIdx.x;  // b*T + t
  const int t   = row % T_;
  const int tok = idx[row];
  for (int i = threadIdx.x; i < E_; i += 256)
    x[(size_t)row * E_ + i] =
        wte[(size_t)tok * E_ + i] + wpe[(size_t)t * E_ + i];
}

// LayerNorm; output written in f16 (sole consumers are GEMM A-operands).
__global__ __launch_bounds__(256) void layernorm_kernel(
    const float* __restrict__ x, const float* __restrict__ w,
    const float* __restrict__ bia, _Float16* __restrict__ out) {
  __shared__ float red[256];
  const int row = blockIdx.x;
  const float* xr = x + (size_t)row * E_;
  float s = 0.f, ss = 0.f;
  for (int i = threadIdx.x; i < E_; i += 256) {
    const float v = xr[i];
    s += v; ss += v * v;
  }
  red[threadIdx.x] = s; __syncthreads();
  for (int o = 128; o > 0; o >>= 1) {
    if (threadIdx.x < o) red[threadIdx.x] += red[threadIdx.x + o];
    __syncthreads();
  }
  const float mean = red[0] / (float)E_;
  __syncthreads();
  red[threadIdx.x] = ss; __syncthreads();
  for (int o = 128; o > 0; o >>= 1) {
    if (threadIdx.x < o) red[threadIdx.x] += red[threadIdx.x + o];
    __syncthreads();
  }
  const float var  = red[0] / (float)E_ - mean * mean;
  const float rstd = rsqrtf(var + 1e-5f);
  _Float16* orow = out + (size_t)row * E_;
  for (int i = threadIdx.x; i < E_; i += 256)
    orow[i] = (_Float16)((xr[i] - mean) * rstd * w[i] + bia[i]);
}

__global__ __launch_bounds__(256) void nll_kernel(
    const float* __restrict__ logits, const int* __restrict__ tgt,
    float* __restrict__ nll, float* __restrict__ valid) {
  __shared__ float red[256];
  const int row = blockIdx.x;
  const float* lr = logits + (size_t)row * V_;
  float mx = -__builtin_inff();
  for (int i = threadIdx.x; i < V_; i += 256) mx = fmaxf(mx, lr[i]);
  red[threadIdx.x] = mx; __syncthreads();
  for (int o = 128; o > 0; o >>= 1) {
    if (threadIdx.x < o)
      red[threadIdx.x] = fmaxf(red[threadIdx.x], red[threadIdx.x + o]);
    __syncthreads();
  }
  mx = red[0]; __syncthreads();
  float s = 0.f;
  for (int i = threadIdx.x; i < V_; i += 256) s += expf(lr[i] - mx);
  red[threadIdx.x] = s; __syncthreads();
  for (int o = 128; o > 0; o >>= 1) {
    if (threadIdx.x < o) red[threadIdx.x] += red[threadIdx.x + o];
    __syncthreads();
  }
  if (threadIdx.x == 0) {
    const float logZ = mx + logf(red[0]);
    const int t = tgt[row];
    const bool ok = (t != -1);
    nll[row]   = ok ? (logZ - lr[t]) : 0.0f;
    valid[row] = ok ? 1.0f : 0.0f;
  }
}

__global__ __launch_bounds__(256) void loss_final_kernel(
    const float* __restrict__ nll, const float* __restrict__ valid,
    float* __restrict__ loss) {
  __shared__ float r1[256], r2[256];
  float s = 0.f, c = 0.f;
  for (int i = threadIdx.x; i < BT_; i += 256) { s += nll[i]; c += valid[i]; }
  r1[threadIdx.x] = s; r2[threadIdx.x] = c; __syncthreads();
  for (int o = 128; o > 0; o >>= 1) {
    if (threadIdx.x < o) {
      r1[threadIdx.x] += r1[threadIdx.x + o];
      r2[threadIdx.x] += r2[threadIdx.x + o];
    }
    __syncthreads();
  }
  if (threadIdx.x == 0) loss[0] = r1[0] / fmaxf(r2[0], 1.0f);
}

// ---------------------------------------------------------------------------
static inline size_t align256(size_t v) { return (v + 255) & ~(size_t)255; }

extern "C" void kernel_launch(void* const* d_in, const int* in_sizes, int n_in,
                              void* d_out, int out_size, void* d_ws,
                              size_t ws_size, hipStream_t stream) {
  (void)in_sizes; (void)n_in; (void)out_size; (void)ws_size;
  const int*   idx       = (const int*)d_in[0];
  const int*   targets   = (const int*)d_in[1];
  const float* wte       = (const float*)d_in[2];
  const float* wpe       = (const float*)d_in[3];
  const float* ln1_w     = (const float*)d_in[4];
  const float* ln1_b     = (const float*)d_in[5];
  const float* attn_qkv  = (const float*)d_in[6];
  const float* attn_proj = (const float*)d_in[7];
  const float* ln2_w     = (const float*)d_in[8];
  const float* ln2_b     = (const float*)d_in[9];
  const float* mlp_fc    = (const float*)d_in[10];
  const float* mlp_proj  = (const float*)d_in[11];
  const float* lnf_w     = (const float*)d_in[12];
  const float* lnf_b     = (const float*)d_in[13];

  float* logits = (float*)d_out;
  float* loss   = logits + (size_t)BT_ * V_;

  // Workspace carve-up (256B aligned slices).
  char* p = (char*)d_ws;
  auto take = [&](size_t bytes) {
    char* r = p;
    p += align256(bytes);
    return r;
  };
  float*    x      = (float*)take((size_t)BT_ * E_ * 4);
  _Float16* h16    = (_Float16*)take((size_t)BT_ * E_ * 2);
  _Float16* qkv16  = (_Float16*)take((size_t)BT_ * 3 * E_ * 2);
  _Float16* y16    = (_Float16*)take((size_t)BT_ * E_ * 2);
  _Float16* m16    = (_Float16*)take((size_t)BT_ * 4 * E_ * 2);
  _Float16* wte16  = (_Float16*)take((size_t)V_ * E_ * 2);
  _Float16* wqkv16 = (_Float16*)take((size_t)L_ * 3 * E_ * E_ * 2);
  _Float16* wprj16 = (_Float16*)take((size_t)L_ * E_ * E_ * 2);
  _Float16* wfc16  = (_Float16*)take((size_t)L_ * 4 * E_ * E_ * 2);
  _Float16* wmp16  = (_Float16*)take((size_t)L_ * E_ * 4 * E_ * 2);
  float*    nllb   = (float*)take((size_t)BT_ * 4);
  float*    valb   = (float*)take((size_t)BT_ * 4);

  // One-time (per call) f16 conversion of all GEMM weights.
  cvt_f32_to_f16<<<2048, 256, 0, stream>>>(wte, wte16, (long)V_ * E_);
  cvt_f32_to_f16<<<2048, 256, 0, stream>>>(attn_qkv, wqkv16,
                                           (long)L_ * 3 * E_ * E_);
  cvt_f32_to_f16<<<2048, 256, 0, stream>>>(attn_proj, wprj16,
                                           (long)L_ * E_ * E_);
  cvt_f32_to_f16<<<2048, 256, 0, stream>>>(mlp_fc, wfc16,
                                           (long)L_ * 4 * E_ * E_);
  cvt_f32_to_f16<<<2048, 256, 0, stream>>>(mlp_proj, wmp16,
                                           (long)L_ * E_ * 4 * E_);

  embed_kernel<<<BT_, 256, 0, stream>>>(idx, wte, wpe, x);

  for (int l = 0; l < L_; ++l) {
    layernorm_kernel<<<BT_, 256, 0, stream>>>(
        x, ln1_w + (size_t)l * E_, ln1_b + (size_t)l * E_, h16);
    gemm_wmma_async<3><<<dim3(BT_ / BM, (3 * E_) / BN), 256, 0, stream>>>(
        h16, wqkv16 + (size_t)l * 3 * E_ * E_, nullptr, nullptr, qkv16,
        BT_, 3 * E_, E_);
    attn_flash<<<dim3(T_ / 64, B_ * H_), 128, 0, stream>>>(qkv16, y16);
    gemm_wmma_async<1><<<dim3(BT_ / BM, E_ / BN), 256, 0, stream>>>(
        y16, wprj16 + (size_t)l * E_ * E_, x, x, nullptr, BT_, E_, E_);
    layernorm_kernel<<<BT_, 256, 0, stream>>>(
        x, ln2_w + (size_t)l * E_, ln2_b + (size_t)l * E_, h16);
    gemm_wmma_async<2><<<dim3(BT_ / BM, (4 * E_) / BN), 256, 0, stream>>>(
        h16, wfc16 + (size_t)l * 4 * E_ * E_, nullptr, nullptr, m16,
        BT_, 4 * E_, E_);
    gemm_wmma_async<1><<<dim3(BT_ / BM, E_ / BN), 256, 0, stream>>>(
        m16, wmp16 + (size_t)l * E_ * 4 * E_, x, x, nullptr, BT_, E_, 4 * E_);
  }

  layernorm_kernel<<<BT_, 256, 0, stream>>>(x, lnf_w, lnf_b, h16);
  gemm_wmma_async<0><<<dim3(BT_ / BM, (V_ + BN - 1) / BN), 256, 0, stream>>>(
      h16, wte16, nullptr, logits, nullptr, BT_, V_, E_);
  nll_kernel<<<BT_, 256, 0, stream>>>(logits, targets, nllb, valb);
  loss_final_kernel<<<1, 256, 0, stream>>>(nllb, valb, loss);
}
